// MultiheadAttention_47038481826236
// MI455X (gfx1250) — matrix operand order repaired and checked by reading, not compiled, not run on hardware
//
#include <hip/hip_runtime.h>
#include <stdint.h>

// ---------------------------------------------------------------------------
// CDNA5 (gfx1250) multihead attention: bf16 WMMA + async global->LDS staging.
// ---------------------------------------------------------------------------

typedef __attribute__((ext_vector_type(16))) __bf16 v16bf;
typedef __attribute__((ext_vector_type(8)))  float  v8f;

union Frag16 {
  uint4          q[2];
  unsigned short s[16];
  v16bf          v;
};

__device__ __forceinline__ unsigned short f32_to_bf16(float f) {
  union { float f; uint32_t u; } x;
  x.f = f;
  uint32_t u = x.u;
  // round-to-nearest-even
  uint32_t r = (u + 0x7FFFu + ((u >> 16) & 1u)) >> 16;
  return (unsigned short)r;
}

__device__ __forceinline__ v8f wmma_bf16(const Frag16& a, const Frag16& b, v8f c) {
  // D = A(16x32 bf16) * B(32x16 bf16) + C(16x16 f32)
  return __builtin_amdgcn_wmma_f32_16x16x32_bf16(
      /*neg_a=*/false, a.v, /*neg_b=*/false, b.v,
      /*c_mod=*/(short)0, c, /*reuse_a=*/false, /*reuse_b=*/false);
}

// LDS byte offset of a __shared__ location: addrspacecast(3->flat) keeps the
// LDS offset in the low 32 bits of the flat address.
__device__ __forceinline__ uint32_t lds_off32(const void* p) {
  return (uint32_t)(uint64_t)(uintptr_t)p;
}

// GLOBAL_LOAD_ASYNC_TO_LDS_B128: per-lane 16B global -> LDS, ASYNCcnt tracked.
__device__ __forceinline__ void async_copy_b128(uint32_t lds_addr, const void* gptr) {
  asm volatile("global_load_async_to_lds_b128 %0, %1, off"
               :
               : "v"(lds_addr), "v"(gptr)
               : "memory");
}

__device__ __forceinline__ void wait_async0() {
  asm volatile("s_wait_asynccnt 0" ::: "memory");
}

// convert 16 consecutive f32 -> 16 bf16, store as 2x uint4 (32B) to LDS
__device__ __forceinline__ void cvt16_store(unsigned short* dst, const float4* src) {
  float4 f0 = src[0], f1 = src[1], f2 = src[2], f3 = src[3];
  union { unsigned short s[16]; uint4 q[2]; } p;
  p.s[0]  = f32_to_bf16(f0.x); p.s[1]  = f32_to_bf16(f0.y);
  p.s[2]  = f32_to_bf16(f0.z); p.s[3]  = f32_to_bf16(f0.w);
  p.s[4]  = f32_to_bf16(f1.x); p.s[5]  = f32_to_bf16(f1.y);
  p.s[6]  = f32_to_bf16(f1.z); p.s[7]  = f32_to_bf16(f1.w);
  p.s[8]  = f32_to_bf16(f2.x); p.s[9]  = f32_to_bf16(f2.y);
  p.s[10] = f32_to_bf16(f2.z); p.s[11] = f32_to_bf16(f2.w);
  p.s[12] = f32_to_bf16(f3.x); p.s[13] = f32_to_bf16(f3.y);
  p.s[14] = f32_to_bf16(f3.z); p.s[15] = f32_to_bf16(f3.w);
  uint4* d = (uint4*)dst;
  d[0] = p.q[0];
  d[1] = p.q[1];
}

// ---------------------------------------------------------------------------
// GEMM: C[M,N] = X[M,K] @ W[N,K]^T + bias[N]
// 128 threads (4 waves); block computes a 128(M) x 64(N) tile.
// Each wave: 2 M-tiles x 4 N-tiles = 8 WMMA per 32-deep K chunk.
// IN_BF16:  X is bf16 row-major -> staged with async global->LDS copies.
// OUT_HS :  store bf16 in head-split layout [B*H][S][64]; else f32 [M][N].
// ---------------------------------------------------------------------------
template <bool IN_BF16, bool OUT_HS>
__global__ __launch_bounds__(128) void gemm_wmma(const void* __restrict__ Xv,
                                                 const float* __restrict__ W,
                                                 const float* __restrict__ bias,
                                                 void* __restrict__ Ov,
                                                 int M, int N, int K) {
  __shared__ __align__(16) unsigned short As[128 * 32];  // 8 KB
  __shared__ __align__(16) unsigned short Bs[64 * 32];   // 4 KB

  const int tid   = threadIdx.x;
  const int lane  = tid & 31;
  const int wave  = tid >> 5;
  const int n0    = blockIdx.x * 64;
  const int m0    = blockIdx.y * 128;
  const int nlane = lane & 15;
  const int mbase = (lane < 16) ? 0 : 8;   // C-fragment row base
  const int kb    = (lane < 16) ? 0 : 8;   // A-fragment K base
  const int bo    = (lane < 16) ? 0 : 8;   // B-fragment K base

  const int brow = tid >> 1;        // B staging row 0..63
  const int bcb  = (tid & 1) * 16;  // 0 or 16

  v8f c[2][4] = {};

  for (int kc = 0; kc < K; kc += 32) {
    // ---- stage A tile (128 x 32): one full row per thread ----
    if (IN_BF16) {
      const unsigned short* X = (const unsigned short*)Xv;
      const unsigned short* xr = X + (size_t)(m0 + tid) * K + kc;
      const uint32_t la = lds_off32(As + tid * 32);
#pragma unroll
      for (int i = 0; i < 4; ++i) async_copy_b128(la + i * 16, xr + i * 8);
    } else {
      const float* X = (const float*)Xv;
      const float* xr = X + (size_t)(m0 + tid) * K + kc;
      cvt16_store(As + tid * 32,      (const float4*)xr);
      cvt16_store(As + tid * 32 + 16, (const float4*)(xr + 16));
    }
    // ---- stage B tile (W sub-block, already [n][k] row-major) ----
    cvt16_store(Bs + brow * 32 + bcb,
                (const float4*)(W + (size_t)(n0 + brow) * K + kc + bcb));

    // prefetch next K chunk (global_prefetch_b8)
    if (kc + 32 < K) {
      if (!IN_BF16)
        __builtin_prefetch((const float*)Xv + (size_t)(m0 + tid) * K + kc + 32, 0, 1);
      __builtin_prefetch(W + (size_t)(n0 + brow) * K + kc + 32 + bcb, 0, 1);
    }
    if (IN_BF16) wait_async0();
    __syncthreads();

    // A fragments: this wave's two 16-row tiles
    Frag16 a[2];
#pragma unroll
    for (int mt = 0; mt < 2; ++mt) {
      const unsigned short* arow = As + (wave * 32 + mt * 16 + nlane) * 32;
      a[mt].q[0] = *(const uint4*)(arow + kb);
      a[mt].q[1] = *(const uint4*)(arow + kb + 16);
    }

#pragma unroll
    for (int nt = 0; nt < 4; ++nt) {
      Frag16 b;
      const unsigned short* brw = Bs + (nt * 16 + nlane) * 32 + bo;
      b.q[0] = *(const uint4*)(brw);
      b.q[1] = *(const uint4*)(brw + 16);
#pragma unroll
      for (int mt = 0; mt < 2; ++mt)
        c[mt][nt] = wmma_bf16(a[mt], b, c[mt][nt]);
    }
    __syncthreads();
  }

  // ---- epilogue: bias + store ----
#pragma unroll
  for (int mt = 0; mt < 2; ++mt) {
#pragma unroll
    for (int nt = 0; nt < 4; ++nt) {
      const int n    = n0 + nt * 16 + nlane;
      const float bv = bias[n];
#pragma unroll
      for (int r = 0; r < 8; ++r) {
        const int m   = m0 + wave * 32 + mt * 16 + mbase + r;
        const float v = c[mt][nt][r] + bv;
        if (OUT_HS) {
          unsigned short* O = (unsigned short*)Ov;
          const int h = n >> 6, dk = n & 63;
          const int bi = m >> 11, s = m & 2047;
          O[(((size_t)(bi * 12 + h) * 2048 + s) << 6) + dk] = f32_to_bf16(v);
        } else {
          float* O = (float*)Ov;
          O[(size_t)m * N + n] = v;
        }
      }
    }
  }
}

// ---------------------------------------------------------------------------
// Flash attention: 128 threads = 4 waves per block; each wave owns one
// 16-query block, all four share one K/V chunk staged in LDS (4x reuse).
// Q and K tiles staged via async global->LDS; V staged with a register
// transpose. Per wave per 32-key chunk: 4 WMMA (QK^T) + 4 WMMA (PV).
// ---------------------------------------------------------------------------
__global__ __launch_bounds__(128) void flash_attn(const unsigned short* __restrict__ Qh,
                                                  const unsigned short* __restrict__ Kh,
                                                  const unsigned short* __restrict__ Vh,
                                                  unsigned short* __restrict__ concat) {
  const int tid  = threadIdx.x;
  const int lane = tid & 31;
  const int wave = tid >> 5;
  const int bh   = blockIdx.y;                 // 0..23
  const int qb   = blockIdx.x * 4 + wave;      // 0..127
  const int bi   = bh / 12;
  const int h    = bh % 12;

  const unsigned short* Q  = Qh + (size_t)bh * 2048 * 64;
  const unsigned short* Kp = Kh + (size_t)bh * 2048 * 64;
  const unsigned short* Vp = Vh + (size_t)bh * 2048 * 64;

  __shared__ __align__(16) unsigned short Qs[4][16 * 64];  // 8 KB (per-wave)
  __shared__ __align__(16) unsigned short Ks[32 * 64];     // 4 KB (shared)  [key][dk]
  __shared__ __align__(16) unsigned short Vs[64 * 40];     // 5 KB (shared)  [dk][key] pad 40
  __shared__ __align__(16) float          Ss[4][16 * 32];  // 8 KB (per-wave)
  __shared__ __align__(16) unsigned short Ps[4][16 * 32];  // 4 KB (per-wave)
  __shared__ float rowM[4][16], rowL[4][16], rowScale[4][16];

  const int nlane = lane & 15;
  const int mbase = (lane < 16) ? 0 : 8;
  const int kb    = (lane < 16) ? 0 : 8;
  const int bo    = (lane < 16) ? 0 : 8;

  // K/V staging map: 4 threads per key row, 16 dk each
  const int krow = tid >> 2;        // 0..31
  const int kcb  = (tid & 3) * 16;  // 0,16,32,48

  // load this wave's Q block 16x64 via async copies (one half-row per thread)
  {
    const int r = lane >> 1, cb = (lane & 1) * 32;
    const unsigned short* src = Q + (size_t)(qb * 16 + r) * 64 + cb;
    const uint32_t la = lds_off32(Qs[wave] + r * 64 + cb);
#pragma unroll
    for (int i = 0; i < 4; ++i) async_copy_b128(la + i * 16, src + i * 8);
  }
  if (lane < 16) { rowM[wave][lane] = -3.0e38f; rowL[wave][lane] = 0.0f; }

  v8f o[4] = {};

  for (int kc = 0; kc < 2048; kc += 32) {
    // ---- stage K chunk (32x64 row-major = B-tile [n][k]) via async ----
    {
      const unsigned short* src = Kp + (size_t)(kc + krow) * 64 + kcb;
      const uint32_t la = lds_off32(Ks + krow * 64 + kcb);
      async_copy_b128(la, src);
      async_copy_b128(la + 16, src + 8);
    }
    // ---- stage V chunk transposed: Vs[dk][key], register transpose ----
    {
      const uint4* src = (const uint4*)(Vp + (size_t)(kc + krow) * 64 + kcb);
      uint4 vb[2];
      vb[0] = src[0];
      vb[1] = src[1];
      const unsigned short* vs = (const unsigned short*)vb;
#pragma unroll
      for (int i = 0; i < 16; ++i) Vs[(kcb + i) * 40 + krow] = vs[i];
    }
    if (kc + 32 < 2048) {
      __builtin_prefetch(Kp + (size_t)(kc + 32 + krow) * 64 + kcb, 0, 1);
      __builtin_prefetch(Vp + (size_t)(kc + 32 + krow) * 64 + kcb, 0, 1);
    }
    wait_async0();
    __syncthreads();

    // ---- scores S = Q K^T (16 x 32), two 16-col tiles, K-dim 64 = 2 steps ----
    Frag16 aq0, aq1;
    {
      const unsigned short* arow = Qs[wave] + nlane * 64;
      aq0.q[0] = *(const uint4*)(arow + kb);
      aq0.q[1] = *(const uint4*)(arow + kb + 16);
      aq1.q[0] = *(const uint4*)(arow + 32 + kb);
      aq1.q[1] = *(const uint4*)(arow + 32 + kb + 16);
    }
#pragma unroll
    for (int nt = 0; nt < 2; ++nt) {
      Frag16 b0, b1;
      const unsigned short* brw = Ks + (nt * 16 + nlane) * 64 + bo;
      b0.q[0] = *(const uint4*)(brw);
      b0.q[1] = *(const uint4*)(brw + 16);
      b1.q[0] = *(const uint4*)(brw + 32);
      b1.q[1] = *(const uint4*)(brw + 48);
      v8f sc = {};
      sc = wmma_bf16(aq0, b0, sc);
      sc = wmma_bf16(aq1, b1, sc);
#pragma unroll
      for (int r = 0; r < 8; ++r)
        Ss[wave][(mbase + r) * 32 + nt * 16 + nlane] = sc[r];
    }
    // NOTE: Ss/Ps/row stats are wave-private; DS ops from one wave are
    // in-order, so no block barrier is needed between score-store, softmax
    // and the PV matmul.

    // ---- online softmax update (lanes 0..15 of each wave, one row each) ----
    if (lane < 16) {
      const float mold = rowM[wave][lane];
      float mnew = mold;
      float srow[32];
#pragma unroll
      for (int j = 0; j < 32; ++j) {
        srow[j] = Ss[wave][lane * 32 + j] * 0.125f;  // 1/sqrt(64)
        mnew = fmaxf(mnew, srow[j]);
      }
      const float scale = __expf(mold - mnew);
      float l = rowL[wave][lane] * scale;
#pragma unroll
      for (int j = 0; j < 32; ++j) {
        const float p = __expf(srow[j] - mnew);
        l += p;
        Ps[wave][lane * 32 + j] = f32_to_bf16(p);
      }
      rowM[wave][lane] = mnew;
      rowL[wave][lane] = l;
      rowScale[wave][lane] = scale;
    }

    // ---- rescale running output accumulators ----
    float fac[8];
#pragma unroll
    for (int r = 0; r < 8; ++r) fac[r] = rowScale[wave][mbase + r];
#pragma unroll
    for (int f = 0; f < 4; ++f)
#pragma unroll
      for (int r = 0; r < 8; ++r) o[f][r] *= fac[r];

    // ---- O += P @ V  (A = Ps 16x32, B = Vs[dk][key] per 16-dk tile) ----
    Frag16 ap;
    {
      const unsigned short* arow = Ps[wave] + nlane * 32;
      ap.q[0] = *(const uint4*)(arow + kb);
      ap.q[1] = *(const uint4*)(arow + kb + 16);
    }
#pragma unroll
    for (int f = 0; f < 4; ++f) {
      Frag16 b;
      const unsigned short* brw = Vs + (f * 16 + nlane) * 40 + bo;
      b.q[0] = *(const uint4*)(brw);
      b.q[1] = *(const uint4*)(brw + 16);
      o[f] = wmma_bf16(ap, b, o[f]);
    }
    __syncthreads();  // protect shared Ks/Vs before next chunk's staging
  }

  // ---- finalize: divide by row sums, store bf16 concat [B*S][768] ----
  if (lane < 16) rowScale[wave][lane] = 1.0f / rowL[wave][lane];
  float fac[8];
#pragma unroll
  for (int r = 0; r < 8; ++r) fac[r] = rowScale[wave][mbase + r];
#pragma unroll
  for (int f = 0; f < 4; ++f) {
    const int col = h * 64 + f * 16 + nlane;
#pragma unroll
    for (int r = 0; r < 8; ++r) {
      const int s = qb * 16 + mbase + r;
      concat[((size_t)(bi * 2048 + s)) * 768 + col] = f32_to_bf16(o[f][r] * fac[r]);
    }
  }
}

// ---------------------------------------------------------------------------
// kernel_launch
// ---------------------------------------------------------------------------
extern "C" void kernel_launch(void* const* d_in, const int* in_sizes, int n_in,
                              void* d_out, int out_size, void* d_ws, size_t ws_size,
                              hipStream_t stream) {
  (void)in_sizes; (void)n_in; (void)out_size; (void)ws_size;

  const float* query = (const float*)d_in[0];
  const float* key   = (const float*)d_in[1];
  const float* value = (const float*)d_in[2];
  const float* Wq = (const float*)d_in[3];
  const float* bq = (const float*)d_in[4];
  const float* Wk = (const float*)d_in[5];
  const float* bk = (const float*)d_in[6];
  const float* Wv = (const float*)d_in[7];
  const float* bv = (const float*)d_in[8];
  const float* Wo = (const float*)d_in[9];
  const float* bo = (const float*)d_in[10];

  // workspace: Q,K,V head-split bf16 [24][2048][64], concat bf16 [4096][768]
  const size_t hs_elems = (size_t)24 * 2048 * 64;  // 3,145,728
  unsigned short* Qh = (unsigned short*)d_ws;
  unsigned short* Kh = Qh + hs_elems;
  unsigned short* Vh = Kh + hs_elems;
  unsigned short* Cc = Vh + hs_elems;  // 4096*768

  dim3 gemm_grid(12, 32);  // N/64, M/128
  dim3 gemm_blk(128);

  gemm_wmma<false, true><<<gemm_grid, gemm_blk, 0, stream>>>(query, Wq, bq, Qh, 4096, 768, 768);
  gemm_wmma<false, true><<<gemm_grid, gemm_blk, 0, stream>>>(key,   Wk, bk, Kh, 4096, 768, 768);
  gemm_wmma<false, true><<<gemm_grid, gemm_blk, 0, stream>>>(value, Wv, bv, Vh, 4096, 768, 768);

  flash_attn<<<dim3(32, 24), dim3(128), 0, stream>>>(Qh, Kh, Vh, Cc);

  gemm_wmma<true, false><<<gemm_grid, gemm_blk, 0, stream>>>(Cc, Wo, bo, d_out, 4096, 768, 768);
}
